// EdgeBlock_8203387535865
// MI455X (gfx1250) — compile-verified, matrix operand order checked by reading, not run on hardware
//
#include <hip/hip_runtime.h>
#include <hip/hip_bf16.h>

// ---------------------------------------------------------------------------
// EdgeConv attention block for MI455X (gfx1250), wave32 + WMMA + TDM.
// prep -> sqnorm -> dist(WMMA f32) -> topk(TDM->LDS) -> gather ->
// gemm_h -> gemm_w -> softmax -> gemm_cmix -> gemm_f -> gemm_out
// GEMMs: LDS-staged transposed B panels (ds_load_b128 fragments),
// b128 A-fragment loads, v_wmma_f32_16x16x32_bf16 accumulate.
// ---------------------------------------------------------------------------

typedef __bf16 bf16_t;
typedef __attribute__((ext_vector_type(16))) __bf16 v16bf;
typedef __attribute__((ext_vector_type(8)))  __bf16 v8bf;
typedef __attribute__((ext_vector_type(8)))  float  v8f;
typedef __attribute__((ext_vector_type(2)))  float  v2f;
typedef __attribute__((ext_vector_type(4)))  unsigned int v4u;
typedef __attribute__((ext_vector_type(8)))  int v8i;
typedef __attribute__((ext_vector_type(4)))  int v4i;

#define NB    4
#define CIN   128
#define NP    2048
#define KNN   20
#define FH    128
#define FO    256
#define MCOL  (NP * KNN)        // 40960 edge columns per batch
#define LRELU_NEG 0.01f
#define BN_EPS    1e-5f
#define DIST_INF  3.0e38f

#define KDEPTH 128              // K-depth of one staged LDS panel
#define KPAD   136              // 128 + 8 elems pad; 136*2B = 272B = 17*16B

__device__ __forceinline__ float lrelu(float v) {
  return v >= 0.f ? v : LRELU_NEG * v;
}

// ---------------------------------------------------------------------------
// Accumulate a 16x16 f32 tile over a 128-deep K panel.
// A: global, row-major, lane row = rowBase+li, two contiguous 8xbf16 chunks
//    per fragment (global_load_b128).  B: LDS, transposed [col][k] panel,
//    two contiguous 8xbf16 chunks (ds_load_b128).
// Element map (ISA 7.12.2): e -> K = (e&8)*2 + hi*8 + (e&7).
// ---------------------------------------------------------------------------
__device__ __forceinline__ v8f acc_k128(const bf16_t* aRow, const bf16_t* bRow,
                                        int hi, v8f acc) {
#pragma unroll
  for (int k = 0; k < KDEPTH; k += 32) {
    v8bf aLo = *(const v8bf*)(aRow + k + hi * 8);
    v8bf aHi = *(const v8bf*)(aRow + k + 16 + hi * 8);
    v8bf bLo = *(const v8bf*)(bRow + k + hi * 8);
    v8bf bHi = *(const v8bf*)(bRow + k + 16 + hi * 8);
    v16bf a  = __builtin_shufflevector(aLo, aHi, 0,1,2,3,4,5,6,7,8,9,10,11,12,13,14,15);
    v16bf bb = __builtin_shufflevector(bLo, bHi, 0,1,2,3,4,5,6,7,8,9,10,11,12,13,14,15);
    acc = __builtin_amdgcn_wmma_f32_16x16x32_bf16(false, a, false, bb,
                                                  (short)0, acc, false, false);
  }
  return acc;
}

// ---------------------------------------------------------------------------
// 0) Weight conversion to bf16 + BN folding:  y = s[m]*conv + b'[m]
// ---------------------------------------------------------------------------
__global__ void prep_kernel(
    const float* W1, const float* b1, const float* g1, const float* be1,
    const float* m1, const float* v1,
    const float* W2, const float* b2, const float* g2, const float* be2,
    const float* m2, const float* v2,
    const float* Wx, const float* bx, const float* gx, const float* bex,
    const float* mx, const float* vx,
    const float* Wo,
    bf16_t* w1b, bf16_t* w2b, bf16_t* wxcb, bf16_t* wxdb, bf16_t* wob,
    float* s1, float* bb1, float* s2, float* bb2, float* sx, float* bbx) {
  int t = blockIdx.x * blockDim.x + threadIdx.x;
  if (t < FH * CIN) w1b[t] = (bf16_t)W1[t];
  if (t < FO * FH)  w2b[t] = (bf16_t)W2[t];
  if (t < FO * CIN) {
    int o = t / CIN, c = t % CIN;
    wxcb[t] = (bf16_t)Wx[o * 2 * CIN + c];          // central half
    wxdb[t] = (bf16_t)Wx[o * 2 * CIN + CIN + c];    // (nbr-central) half
  }
  if (t < FO * FO * KNN) wob[t] = (bf16_t)Wo[t];    // (o, c*20+j) row-major
  if (t < FH) {
    float inv = g1[t] * rsqrtf(v1[t] + BN_EPS);
    s1[t] = inv;  bb1[t] = b1[t] * inv + be1[t] - m1[t] * inv;
  }
  if (t < FO) {
    float inv = g2[t] * rsqrtf(v2[t] + BN_EPS);
    s2[t] = inv;  bb2[t] = b2[t] * inv + be2[t] - m2[t] * inv;
    float invx = gx[t] * rsqrtf(vx[t] + BN_EPS);
    sx[t] = invx; bbx[t] = bx[t] * invx + bex[t] - mx[t] * invx;
  }
}

// ---------------------------------------------------------------------------
// 1) Row squared norms + x -> bf16 copy (for the cmix GEMM).
// ---------------------------------------------------------------------------
__global__ void sqnorm_xbf_kernel(const float* x, float* sq, bf16_t* xbf) {
  int t = blockIdx.x * blockDim.x + threadIdx.x;
  if (t >= NB * NP) return;
  int b = t / NP, n = t % NP;
  const float* xb = x + (size_t)b * CIN * NP;
  bf16_t* xbb = xbf + (size_t)b * CIN * NP;
  float s = 0.f;
  for (int c = 0; c < CIN; ++c) {
    float v = xb[(size_t)c * NP + n];
    s += v * v;
    xbb[(size_t)c * NP + n] = (bf16_t)v;
  }
  sq[t] = s;
}

// ---------------------------------------------------------------------------
// 2) Pairwise distance tiles via V_WMMA_F32_16X16X4_F32 (f32 keeps top-k
//    ordering exact).  f32 A/B layout: lane = row/col %16, e -> K = e + 2*hi.
// ---------------------------------------------------------------------------
__global__ void dist_kernel(const float* x, const float* sq, float* dist) {
  int lane = threadIdx.x & 31;
  int wave = threadIdx.x >> 5;
  int mTile = blockIdx.x * 4 + wave;
  int nTile = blockIdx.y;
  int b = blockIdx.z;
  const float* xb = x + (size_t)b * CIN * NP;
  int nRowBase = nTile * 16, mColBase = mTile * 16;
  int li = lane & 15, hi = lane >> 4;
  v8f acc = {};
#if __has_builtin(__builtin_amdgcn_wmma_f32_16x16x4_f32)
  for (int c = 0; c < CIN; c += 4) {
    v2f a, bv;
#pragma unroll
    for (int e = 0; e < 2; ++e) {
      int k = c + e + 2 * hi;
      a[e]  = xb[(size_t)k * NP + nRowBase + li];
      bv[e] = xb[(size_t)k * NP + mColBase + li];
    }
    acc = __builtin_amdgcn_wmma_f32_16x16x4_f32(false, a, false, bv,
                                                (short)0, acc, false, false);
  }
#else
  for (int c = 0; c < CIN; ++c) {
    float bv = xb[(size_t)c * NP + mColBase + li];
#pragma unroll
    for (int r = 0; r < 8; ++r)
      acc[r] += xb[(size_t)c * NP + nRowBase + hi * 8 + r] * bv;
  }
#endif
  const float* sqb = sq + b * NP;
  float sqm = sqb[mColBase + li];
#pragma unroll
  for (int r = 0; r < 8; ++r) {
    int nRow = nRowBase + hi * 8 + r;
    int mCol = mColBase + li;
    float d = sqb[nRow] + sqm - 2.0f * acc[r];
    if (nRow == mCol) d = DIST_INF;   // exclude self (reference drops idx 0)
    dist[((size_t)b * NP + nRow) * NP + mCol] = d;
  }
}

// ---------------------------------------------------------------------------
// 3) Top-20 smallest per row. One wave32 per row; row pulled into LDS by the
//    Tensor Data Mover (8KB 1-row tile, D# per CDNA5 ISA), waited
//    with s_wait_tensorcnt; lanes own slots i === lane (mod 32); butterfly
//    min-reduce via shfl_xor.
// ---------------------------------------------------------------------------
__global__ void topk_kernel(const float* dist, int* idx) {
  __shared__ __align__(16) float sd[4 * NP];     // 4 rows/block, 32 KB
  int wave = threadIdx.x >> 5, lane = threadIdx.x & 31;
  long row = (long)blockIdx.x * 4 + wave;        // [0, NB*NP)
  float* my = sd + (size_t)wave * NP;
  const float* dr = dist + (size_t)row * NP;
#if __has_builtin(__builtin_amdgcn_tensor_load_to_lds) && \
    __has_builtin(__builtin_amdgcn_s_wait_tensorcnt)
  {
    unsigned lds_off = (unsigned)(uintptr_t)my;        // low 32b = LDS offset
    unsigned long long ga = (unsigned long long)(uintptr_t)dr;
    // D# group0: count=1 | lds_addr | global_addr[56:0] | type=2
    v4u g0 = { 1u, lds_off, (unsigned)(ga & 0xFFFFFFFFu),
               (unsigned)((ga >> 32) & 0x01FFFFFFu) | (2u << 30) };
    // D# group1: data_size=4B; tensor_dim0=2048; tile_dim0=2048 (1 row);
    //            tensor_dim0_stride=2048; no multicast/pad/iterate.
    v8i g1 = { (int)(2u << 16),        // [17:16] data_size=2 -> 4 bytes
               (int)(2048u << 16),     // [63:48] tensor_dim0 lo16
               0,                      // tensor_dim0 hi / tensor_dim1 lo
               (int)(2048u << 16),     // [127:112] tile_dim0 = 2048
               0,                      // tile_dim1 = 0 (unused)
               2048,                   // [191:160] tensor_dim0_stride lo32
               0, 0 };
    v4i gz4 = { 0, 0, 0, 0 };
    v8i gz8 = { 0, 0, 0, 0, 0, 0, 0, 0 };
    // amdgpu-toolchain (clang-23) 6-arg form: (g0, g1, g2, g3, g4, cpol)
    __builtin_amdgcn_tensor_load_to_lds(g0, g1, gz4, gz4, gz8, 0);
    __builtin_amdgcn_s_wait_tensorcnt(0);
  }
#else
  for (int i = lane; i < NP; i += 32) my[i] = dr[i];
#endif
  for (int j = 0; j < KNN; ++j) {
    float bestd = DIST_INF;
    int besti = 0x7fffffff;
    for (int i = lane; i < NP; i += 32) {
      float d = my[i];
      if (d < bestd) { bestd = d; besti = i; }
    }
#pragma unroll
    for (int off = 16; off; off >>= 1) {
      float od = __shfl_xor(bestd, off, 32);
      int   oi = __shfl_xor(besti, off, 32);
      if (od < bestd || (od == bestd && oi < besti)) { bestd = od; besti = oi; }
    }
    if (lane == 0) idx[(size_t)row * KNN + j] = besti;
    if (besti < NP && (besti & 31) == lane) my[besti] = DIST_INF;
  }
}

// ---------------------------------------------------------------------------
// 4) Edge gather: ediff[b][c][n*20+j] = x[b][c][idx] - x[b][c][n]  (bf16)
// ---------------------------------------------------------------------------
__global__ void edge_gather_kernel(const float* x, const int* idx, bf16_t* ediff) {
  long t = (long)blockIdx.x * blockDim.x + threadIdx.x;
  if (t >= (long)NB * CIN * NP * KNN) return;
  int j = (int)(t % KNN);
  long r = t / KNN;
  int n = (int)(r % NP); r /= NP;
  int c = (int)(r % CIN); r /= CIN;
  int b = (int)r;
  int nb = idx[((size_t)b * NP + n) * KNN + j];
  const float* xc = x + ((size_t)b * CIN + c) * NP;
  ediff[((size_t)(b * CIN + c)) * MCOL + (size_t)n * KNN + j] =
      (bf16_t)(xc[nb] - xc[n]);
}

// ---------------------------------------------------------------------------
// Staged GEMM bodies: block = 128 threads (4 waves), 16 rows x 64 cols tile.
// Stage B panel [64 cols][128 k] transposed into LDS, then each wave owns a
// 16-col slice; fragments are 2x ds_load_b128 (B) + 2x global_load_b128 (A).
// ---------------------------------------------------------------------------
#define GEMM_PROLOG(ACT, LDB, MPER)                                            \
  __shared__ __align__(16) bf16_t tileB[64][KPAD];                             \
  int lane = threadIdx.x & 31, wave = threadIdx.x >> 5;                        \
  int li = lane & 15, hi = lane >> 4;                                          \
  long tcolBase = (long)blockIdx.x * 64;                                       \
  int b = (int)(tcolBase / (MPER));                                            \
  long colBase = tcolBase - (long)b * (MPER);                                  \
  int rowBase = blockIdx.y * 16;                                               \
  for (int s = threadIdx.x; s < 64 * KDEPTH; s += 128) {                       \
    int c = s & 63, k = s >> 6;                                                \
    tileB[c][k] = (ACT)[(size_t)k * (LDB) + colBase + c];                      \
  }                                                                            \
  __syncthreads();                                                             \
  v8f acc = {};

// ---------------------------------------------------------------------------
// 5) h = lrelu(bn(W1 @ ediff))   [128 x 163840]
// ---------------------------------------------------------------------------
__global__ void __launch_bounds__(128)
gemm_h_kernel(const bf16_t* w1b, const bf16_t* ediff,
              const float* s1, const float* bb1, bf16_t* h) {
  const bf16_t* Act0 = ediff + ((size_t)((long)blockIdx.x * 64 / MCOL)) * CIN * MCOL;
  GEMM_PROLOG(Act0, MCOL, MCOL)
  acc = acc_k128(w1b + (size_t)(rowBase + li) * CIN, &tileB[wave * 16 + li][0], hi, acc);
  long col = colBase + wave * 16 + li;
#pragma unroll
  for (int r = 0; r < 8; ++r) {
    int m = rowBase + hi * 8 + r;
    float v = lrelu(s1[m] * acc[r] + bb1[m]);
    h[((size_t)(b * FH + m)) * MCOL + col] = (bf16_t)v;
  }
}

// ---------------------------------------------------------------------------
// 6) wpre = lrelu(bn(W2 @ h))   [256 x 163840] f32 (softmax needs f32)
// ---------------------------------------------------------------------------
__global__ void __launch_bounds__(128)
gemm_w_kernel(const bf16_t* w2b, const bf16_t* h,
              const float* s2, const float* bb2, float* wpre) {
  const bf16_t* Act0 = h + ((size_t)((long)blockIdx.x * 64 / MCOL)) * FH * MCOL;
  GEMM_PROLOG(Act0, MCOL, MCOL)
  acc = acc_k128(w2b + (size_t)(rowBase + li) * FH, &tileB[wave * 16 + li][0], hi, acc);
  long col = colBase + wave * 16 + li;
#pragma unroll
  for (int r = 0; r < 8; ++r) {
    int m = rowBase + hi * 8 + r;
    wpre[((size_t)(b * FO + m)) * MCOL + col] = lrelu(s2[m] * acc[r] + bb2[m]);
  }
}

// ---------------------------------------------------------------------------
// 7) softmax over the K=20 neighbor axis -> bf16
// ---------------------------------------------------------------------------
__global__ void softmax_kernel(const float* wpre, bf16_t* wsm) {
  long t = (long)blockIdx.x * blockDim.x + threadIdx.x;
  if (t >= (long)NB * FO * NP) return;
  int n = (int)(t % NP);
  long r = t / NP;
  int o = (int)(r % FO);
  int b = (int)(r / FO);
  size_t base = ((size_t)(b * FO + o)) * MCOL + (size_t)n * KNN;
  const float* p = wpre + base;
  float mx = p[0];
#pragma unroll
  for (int j = 1; j < KNN; ++j) mx = fmaxf(mx, p[j]);
  float e[KNN], sum = 0.f;
#pragma unroll
  for (int j = 0; j < KNN; ++j) { e[j] = __expf(p[j] - mx); sum += e[j]; }
  float inv = 1.f / sum;
  bf16_t* q = wsm + base;
#pragma unroll
  for (int j = 0; j < KNN; ++j) q[j] = (bf16_t)(e[j] * inv);
}

// ---------------------------------------------------------------------------
// 8) cmix = Wx_c @ x  (central half, broadcast over j later)  [256 x 8192]
// ---------------------------------------------------------------------------
__global__ void __launch_bounds__(128)
gemm_cmix_kernel(const bf16_t* wxcb, const bf16_t* xbf, float* cmix) {
  const bf16_t* Act0 = xbf + ((size_t)((long)blockIdx.x * 64 / NP)) * CIN * NP;
  GEMM_PROLOG(Act0, NP, NP)
  acc = acc_k128(wxcb + (size_t)(rowBase + li) * CIN, &tileB[wave * 16 + li][0], hi, acc);
  long col = colBase + wave * 16 + li;
#pragma unroll
  for (int r = 0; r < 8; ++r) {
    int m = rowBase + hi * 8 + r;
    cmix[((size_t)(b * FO + m)) * NP + col] = acc[r];
  }
}

// ---------------------------------------------------------------------------
// 9) f = lrelu(bn(Wx_d @ ediff + cmix)) * softmax(w)   -> bf16 fbuf
// ---------------------------------------------------------------------------
__global__ void __launch_bounds__(128)
gemm_f_kernel(const bf16_t* wxdb, const bf16_t* ediff,
              const float* cmix, const float* sx, const float* bbx,
              const bf16_t* wsm, bf16_t* fbuf) {
  const bf16_t* Act0 = ediff + ((size_t)((long)blockIdx.x * 64 / MCOL)) * CIN * MCOL;
  GEMM_PROLOG(Act0, MCOL, MCOL)
  acc = acc_k128(wxdb + (size_t)(rowBase + li) * CIN, &tileB[wave * 16 + li][0], hi, acc);
  long cc = colBase + wave * 16 + li;
  int n = (int)(cc / KNN);
#pragma unroll
  for (int r = 0; r < 8; ++r) {
    int m = rowBase + hi * 8 + r;
    size_t chan = (size_t)(b * FO + m);
    float v = acc[r] + cmix[chan * NP + n];
    v = lrelu(sx[m] * v + bbx[m]);
    v *= (float)wsm[chan * MCOL + cc];
    fbuf[chan * MCOL + cc] = (bf16_t)v;
  }
}

// ---------------------------------------------------------------------------
// 10) out[b][o][n] = sum_{c,j} Wo[o][c*20+j] * f[b][c][n*20+j] + bo[o]
//     256 x 5120 x 8192 GEMM; B gathered (c,j)->fbuf during LDS staging,
//     then identical ds_load_b128 fragment path; 40 staged K slices.
// ---------------------------------------------------------------------------
__global__ void __launch_bounds__(128)
gemm_out_kernel(const bf16_t* wob, const bf16_t* fbuf,
                const float* bo, float* out) {
  __shared__ __align__(16) bf16_t tileB[64][KPAD];
  int lane = threadIdx.x & 31, wave = threadIdx.x >> 5;
  int li = lane & 15, hi = lane >> 4;
  long tcolBase = (long)blockIdx.x * 64;          // [0, NB*NP)
  int b = (int)(tcolBase / NP);
  int nBase = (int)(tcolBase - (long)b * NP);
  int rowBase = blockIdx.y * 16;
  const bf16_t* Fb = fbuf + (size_t)b * FO * MCOL;
  const bf16_t* aRow = wob + (size_t)(rowBase + li) * (FO * KNN);
  const bf16_t* bRow = &tileB[wave * 16 + li][0];
  v8f acc = {};
  for (int ks = 0; ks < FO * KNN; ks += KDEPTH) {
    __syncthreads();                              // panel reuse barrier
    for (int s = threadIdx.x; s < 64 * KDEPTH; s += 128) {
      int c = s & 63, kk = ks + (s >> 6);
      int ch = kk / KNN, j = kk - ch * KNN;
      tileB[c][s >> 6] = Fb[(size_t)ch * MCOL + (size_t)(nBase + c) * KNN + j];
    }
    __syncthreads();
    acc = acc_k128(aRow + ks, bRow, hi, acc);
  }
  int n = nBase + wave * 16 + li;
#pragma unroll
  for (int r = 0; r < 8; ++r) {
    int o = rowBase + hi * 8 + r;
    out[((size_t)(b * FO + o)) * NP + n] = acc[r] + bo[o];
  }
}

// ---------------------------------------------------------------------------
// Host launcher
// ---------------------------------------------------------------------------
extern "C" void kernel_launch(void* const* d_in, const int* in_sizes, int n_in,
                              void* d_out, int out_size, void* d_ws, size_t ws_size,
                              hipStream_t stream) {
  (void)in_sizes; (void)n_in; (void)out_size; (void)ws_size;
  const float* x   = (const float*)d_in[0];
  const float* W1  = (const float*)d_in[1];
  const float* b1  = (const float*)d_in[2];
  const float* g1  = (const float*)d_in[3];
  const float* be1 = (const float*)d_in[4];
  const float* m1  = (const float*)d_in[5];
  const float* v1  = (const float*)d_in[6];
  const float* W2  = (const float*)d_in[7];
  const float* b2  = (const float*)d_in[8];
  const float* g2  = (const float*)d_in[9];
  const float* be2 = (const float*)d_in[10];
  const float* m2  = (const float*)d_in[11];
  const float* v2  = (const float*)d_in[12];
  const float* Wx  = (const float*)d_in[13];
  const float* bx  = (const float*)d_in[14];
  const float* gx  = (const float*)d_in[15];
  const float* bex = (const float*)d_in[16];
  const float* mx  = (const float*)d_in[17];
  const float* vx  = (const float*)d_in[18];
  const float* Wo  = (const float*)d_in[19];
  const float* bo  = (const float*)d_in[20];
  float* out = (float*)d_out;

  char* ws = (char*)d_ws;
  size_t off = 0;
  auto wsAlloc = [&](size_t bytes) -> char* {
    char* p = ws + off;
    off = (off + bytes + 255) & ~(size_t)255;
    return p;
  };
  bf16_t* xbf   = (bf16_t*)wsAlloc((size_t)NB * CIN * NP * 2);
  float*  sq    = (float*) wsAlloc((size_t)NB * NP * 4);
  float*  dist  = (float*) wsAlloc((size_t)NB * NP * NP * 4);       // 64 MB
  int*    idx   = (int*)   wsAlloc((size_t)NB * NP * KNN * 4);
  bf16_t* ediff = (bf16_t*)wsAlloc((size_t)NB * CIN * MCOL * 2);    // 40 MB
  bf16_t* hbuf  = (bf16_t*)wsAlloc((size_t)NB * FH * MCOL * 2);     // 40 MB
  float*  wpre  = (float*) wsAlloc((size_t)NB * FO * MCOL * 4);     // 160 MB
  bf16_t* wsm   = (bf16_t*)wsAlloc((size_t)NB * FO * MCOL * 2);     // 80 MB
  float*  cmix  = (float*) wsAlloc((size_t)NB * FO * NP * 4);       // 8 MB
  bf16_t* fbuf  = (bf16_t*)wpre;  // alias: wpre dead after softmax
  bf16_t* w1b   = (bf16_t*)wsAlloc((size_t)FH * CIN * 2);
  bf16_t* w2b   = (bf16_t*)wsAlloc((size_t)FO * FH * 2);
  bf16_t* wxcb  = (bf16_t*)wsAlloc((size_t)FO * CIN * 2);
  bf16_t* wxdb  = (bf16_t*)wsAlloc((size_t)FO * CIN * 2);
  bf16_t* wob   = (bf16_t*)wsAlloc((size_t)FO * FO * KNN * 2);
  float*  s1    = (float*) wsAlloc(FH * 4);
  float*  bb1   = (float*) wsAlloc(FH * 4);
  float*  s2    = (float*) wsAlloc(FO * 4);
  float*  bb2   = (float*) wsAlloc(FO * 4);
  float*  sx    = (float*) wsAlloc(FO * 4);
  float*  bbx   = (float*) wsAlloc(FO * 4);

  // 0) prep
  prep_kernel<<<(FO * FO * KNN + 255) / 256, 256, 0, stream>>>(
      W1, b1, g1, be1, m1, v1, W2, b2, g2, be2, m2, v2,
      Wx, bx, gx, bex, mx, vx, Wo,
      w1b, w2b, wxcb, wxdb, wob, s1, bb1, s2, bb2, sx, bbx);
  // 1) sqnorm + bf16 copy of x
  sqnorm_xbf_kernel<<<(NB * NP + 255) / 256, 256, 0, stream>>>(x, sq, xbf);
  // 2) distances (f32 WMMA)
  dist_kernel<<<dim3(NP / 16 / 4, NP / 16, NB), 128, 0, stream>>>(x, sq, dist);
  // 3) top-20 (TDM row load)
  topk_kernel<<<NB * NP / 4, 128, 0, stream>>>(dist, idx);
  // 4) edge gather
  {
    long total = (long)NB * CIN * NP * KNN;
    edge_gather_kernel<<<(unsigned)((total + 255) / 256), 256, 0, stream>>>(x, idx, ediff);
  }
  // 5) h = lrelu(bn(W1 @ ediff))
  gemm_h_kernel<<<dim3(NB * MCOL / 64, FH / 16), 128, 0, stream>>>(w1b, ediff, s1, bb1, hbuf);
  // 6) wpre = lrelu(bn(W2 @ h))
  gemm_w_kernel<<<dim3(NB * MCOL / 64, FO / 16), 128, 0, stream>>>(w2b, hbuf, s2, bb2, wpre);
  // 7) softmax over K
  softmax_kernel<<<(unsigned)(((long)NB * FO * NP + 255) / 256), 256, 0, stream>>>(wpre, wsm);
  // 8) cmix = Wx_c @ x
  gemm_cmix_kernel<<<dim3(NB * NP / 64, FO / 16), 128, 0, stream>>>(wxcb, xbf, cmix);
  // 9) f = lrelu(bn(Wx_d @ ediff + cmix)) * wsm
  gemm_f_kernel<<<dim3(NB * MCOL / 64, FO / 16), 128, 0, stream>>>(
      wxdb, ediff, cmix, sx, bbx, wsm, fbuf);
  // 10) out = Wo (.) f + bo
  gemm_out_kernel<<<dim3(NB * NP / 64, FO / 16), 128, 0, stream>>>(wob, fbuf, bo, out);
}
